// xLSTM_63651415327296
// MI455X (gfx1250) — matrix operand order repaired and checked by reading, not compiled, not run on hardware
//
#include <hip/hip_runtime.h>
#include <hip/hip_bf16.h>

// ---------------------------------------------------------------------------
// xLSTM scan for MI455X (gfx1250). Persistent-kernel design:
//   - batch rows are recurrence-independent -> 16 workgroups x 16 batch rows,
//     each runs the full T=512 scan with state in LDS (320KB/WGP).
//   - all matmuls via v_wmma_f32_16x16x32_bf16 (wave32, 16x16 tiles).
//   - weights pre-packed to WMMA B-fragment layout in bf16 (L2 resident).
//   - mLSTM matrix memory C (B*64*64 f32 = 4MB) lives in global ws / L2.
//   - xh = x@W_in precomputed as one WMMA GEMM (bf16, time-major stream).
//   - bf16 converts use native casts -> v_cvt_*bf16* instead of SW RNE.
// ---------------------------------------------------------------------------

#define Bsz   256
#define Tsz   512
#define DINsz 128
#define Hsz   256
#define Ssz   4
#define Dsz   64
#define HDsz  64
#define EPSF  1e-8f

typedef __attribute__((ext_vector_type(16))) __bf16 bf16x16;
typedef __attribute__((ext_vector_type(8)))  __bf16 bf16x8;
typedef __attribute__((ext_vector_type(8)))  float  f32x8;

__device__ __forceinline__ __bf16 f2bf(float f) { return (__bf16)f; }
__device__ __forceinline__ float  bf2f(__bf16 b) { return (float)b; }
__device__ __forceinline__ float sgm(float x) { return 1.0f / (1.0f + __expf(-x)); }

__device__ __forceinline__ f32x8 wmma_bf16(bf16x16 a, bf16x16 b, f32x8 c) {
  return __builtin_amdgcn_wmma_f32_16x16x32_bf16(false, a, false, b, (short)0, c,
                                                 false, false);
}

// A-fragment (16x32 bf16, ISA 7.12.2): lane l row m=l&15; element e ->
// k = kbase + (e>>3)*16 + (l>>4)*8 + (e&7). Two contiguous 16B LDS loads.
__device__ __forceinline__ bf16x16 load_a_lds(const __bf16* row, int kbase, int half8) {
  bf16x8 lo = *(const bf16x8*)(row + kbase + half8);
  bf16x8 hi = *(const bf16x8*)(row + kbase + 16 + half8);
  return __builtin_shufflevector(lo, hi, 0, 1, 2, 3, 4, 5, 6, 7,
                                 8, 9, 10, 11, 12, 13, 14, 15);
}
// B-fragment: pre-packed so each lane's 16 elements are 32 contiguous bytes.
__device__ __forceinline__ bf16x16 load_b_packed(const __bf16* packed, int frag, int lane) {
  return *(const bf16x16*)(packed + (size_t)frag * 512 + (size_t)lane * 16);
}

// ---------------------------------------------------------------------------
// Phase 1: pack fp32 W[K][N] (row-major) -> bf16 WMMA B fragments.
// frag = nt*(K/32) + kt ; lane l: n = nt*16 + (l&15); element e:
// k = kt*32 + (e>>3)*16 + (l>>4)*8 + (e&7).
// ---------------------------------------------------------------------------
__global__ void pack_b_kernel(const float* __restrict__ W, __bf16* __restrict__ out,
                              int K, int N) {
  int idx = blockIdx.x * blockDim.x + threadIdx.x;
  if (idx >= K * N) return;
  int e    = idx & 15;
  int lane = (idx >> 4) & 31;
  int frag = idx >> 9;
  int KT   = K >> 5;
  int nt   = frag / KT;
  int kt   = frag - nt * KT;
  int n    = nt * 16 + (lane & 15);
  int k    = kt * 32 + (e >> 3) * 16 + (lane >> 4) * 8 + (e & 7);
  out[idx] = f2bf(W[(size_t)k * N + n]);
}

// ---------------------------------------------------------------------------
// Phase 2: xh[t][b][:] = x[b][t][:] @ W_in + b_in  (bf16 out, time-major).
// One 16x16 output tile per wave; K=128 (4 WMMA k-steps).
// ---------------------------------------------------------------------------
__global__ void __launch_bounds__(256) gemm_xh_kernel(
    const float* __restrict__ x, const __bf16* __restrict__ pWin,
    const float* __restrict__ b_in, __bf16* __restrict__ xh) {
  const int lane  = threadIdx.x & 31;
  const int wave  = threadIdx.x >> 5;
  const int ln    = lane & 15;
  const int half8 = (lane >> 4) * 8;
  long gid = (long)blockIdx.x * 8 + wave;
  if (gid >= (long)Tsz * 16 * 16) return;
  int nt = gid & 15;
  int bt = (gid >> 4) & 15;
  int t  = (int)(gid >> 8);

  const float* xrow = x + ((size_t)(bt * 16 + ln) * Tsz + t) * DINsz;
  f32x8 acc;
  float bv = b_in[nt * 16 + ln];
#pragma unroll
  for (int v = 0; v < 8; ++v) acc[v] = bv;

#pragma unroll
  for (int kt = 0; kt < 4; ++kt) {
    bf16x16 a;
#pragma unroll
    for (int i = 0; i < 8; ++i) {
      a[i]     = f2bf(xrow[kt * 32 + half8 + i]);
      a[i + 8] = f2bf(xrow[kt * 32 + 16 + half8 + i]);
    }
    bf16x16 b = load_b_packed(pWin, nt * 4 + kt, lane);
    acc = wmma_bf16(a, b, acc);
  }
#pragma unroll
  for (int v = 0; v < 8; ++v) {
    int mm = v + half8;
    xh[((size_t)t * Bsz + bt * 16 + mm) * Hsz + nt * 16 + ln] = f2bf(acc[v]);
  }
}

// ---------------------------------------------------------------------------
// Phase 3: persistent recurrent scan. 16 blocks x 256 threads (8 waves).
// ---------------------------------------------------------------------------
struct RecParams {
  const __bf16* xh;
  float* C1;
  const __bf16 *pWih0, *pWhh0, *pWq, *pWk, *pWv, *pWo, *pWp, *pSubWih, *pSubWhh, *pAttW1;
  const float *s0_bih, *m_bq, *m_bk, *m_bv, *m_Wi, *m_bi, *m_Wf, *m_bf, *m_bo, *m_bp;
  const float *sub_bih, *sub_fbias, *att_b1, *att_W2, *att_b2, *ln_g, *ln_b;
  float* out;
};

#define GST 1040  // gates LDS row stride (floats), 16B-aligned, padded
#define AST 264   // bf16 activation row stride, multiple of 8

__global__ void __launch_bounds__(256, 1) recurrent_kernel(RecParams P) {
  extern __shared__ char smem[];
  float*  gates = (float*)smem;                        // [16][GST]
  __bf16* xtb   = (__bf16*)(gates + 16 * GST);         // [16][AST]
  __bf16* h0b   = xtb + 16 * AST;                      // [16][AST]
  __bf16* x1b   = h0b + 16 * AST;                      // [16][AST]
  __bf16* x2b   = x1b + 16 * AST;                      // [16][AST]
  __bf16* htb   = x2b + 16 * AST;                      // [16][72]
  __bf16* hsb   = htb + 16 * 72;                       // [4*16][72]
  float*  c0    = (float*)(hsb + 64 * 72);             // [16][256]
  float*  n0    = c0 + 16 * 256;                       // [16][256]
  float*  cs    = n0 + 16 * 256;                       // [4*16][64]
  float*  ns    = cs + 64 * 64;                        // [4*16][64]
  float*  hfu   = ns + 64 * 64;                        // [16][64]
  float*  ipre  = hfu + 16 * 64;                       // [16]
  float*  fpre  = ipre + 16;                           // [16]
  float*  i1a   = fpre + 16;                           // [16]
  float*  f1a   = i1a + 16;                            // [16]
  float*  n1a   = f1a + 16;                            // [16]
  float*  score = n1a + 16;                            // [4*16]
  float*  wsm   = score + 64;                          // [4*16]

  const int tid   = threadIdx.x;
  const int lane  = tid & 31;
  const int wave  = tid >> 5;
  const int ln    = lane & 15;
  const int half8 = (lane >> 4) * 8;
  const int bg    = blockIdx.x;  // batch tile

  // ---- init state: h0=0,c0=0,n0=1 ; hs=0,cs=0,ns=1 ; n1=1 ; C=0 ----
  for (int r = tid; r < 16 * 256; r += 256) {
    c0[r] = 0.0f; n0[r] = 1.0f;
    h0b[(r >> 8) * AST + (r & 255)] = f2bf(0.0f);
  }
  for (int r = tid; r < 64 * 64; r += 256) {
    cs[r] = 0.0f; ns[r] = 1.0f;
    hsb[(r >> 6) * 72 + (r & 63)] = f2bf(0.0f);
  }
  if (tid < 16) n1a[tid] = 1.0f;
  {
    float4* Cb = (float4*)(P.C1 + (size_t)bg * 16 * 64 * 64);
    for (int r = tid; r < 16 * 64 * 64 / 4; r += 256) Cb[r] = make_float4(0, 0, 0, 0);
  }
  __syncthreads();

  for (int t = 0; t < Tsz; ++t) {
    const __bf16* xtg = P.xh + ((size_t)t * Bsz + bg * 16) * Hsz;
    // ---- stage xt tile into LDS (32B per thread) + prefetch next t ----
    {
      int m = tid >> 4, c = (tid & 15) * 16;
      const __bf16* s = xtg + (size_t)m * Hsz + c;
      __bf16* d = xtb + m * AST + c;
      *(bf16x8*)d       = *(const bf16x8*)s;
      *(bf16x8*)(d + 8) = *(const bf16x8*)(s + 8);
      if (t + 1 < Tsz) __builtin_prefetch(xtg + Bsz * Hsz + tid * 16, 0, 1);
      if (tid < 16) { ipre[tid] = P.m_bi[0]; fpre[tid] = P.m_bf[0]; }
    }
    __syncthreads();

    // ---- Part A: layer0 gates = xt@Wih + h0@Whh + bih   (64 n-tiles) ----
    for (int j = 0; j < 8; ++j) {
      int nt = wave * 8 + j;
      f32x8 acc; float bv = P.s0_bih[nt * 16 + ln];
#pragma unroll
      for (int v = 0; v < 8; ++v) acc[v] = bv;
      const __bf16* ax = xtb + ln * AST;
      const __bf16* ah = h0b + ln * AST;
#pragma unroll
      for (int kt = 0; kt < 8; ++kt)
        acc = wmma_bf16(load_a_lds(ax, kt * 32, half8),
                        load_b_packed(P.pWih0, nt * 8 + kt, lane), acc);
#pragma unroll
      for (int kt = 0; kt < 8; ++kt)
        acc = wmma_bf16(load_a_lds(ah, kt * 32, half8),
                        load_b_packed(P.pWhh0, nt * 8 + kt, lane), acc);
#pragma unroll
      for (int v = 0; v < 8; ++v) gates[(v + half8) * GST + nt * 16 + ln] = acc[v];
    }
    __syncthreads();

    // ---- A2: sLSTM pointwise, x1 = h0n + xt ----
    for (int r = tid; r < 4096; r += 256) {
      int m = r >> 8, h = r & 255;
      float* g = gates + m * GST;
      float i = __expf(g[h]), f = sgm(g[256 + h]), o = sgm(g[512 + h]);
      float cn = f * c0[m * 256 + h] + i * tanhf(g[768 + h]);
      float nn = f * n0[m * 256 + h] + i;
      float hn = o * tanhf(cn / (nn + EPSF));
      c0[m * 256 + h] = cn; n0[m * 256 + h] = nn;
      h0b[m * AST + h] = f2bf(hn);
      x1b[m * AST + h] = f2bf(hn + bf2f(xtb[m * AST + h]));
    }
    __syncthreads();

    // ---- Part B: o/q/k/v projections (28 tiles) + i/f dot partials ----
    for (int tt = wave; tt < 28; tt += 8) {
      const __bf16* pw; const float* bias; int col0, nl;
      if (tt < 16)      { pw = P.pWo; bias = P.m_bo; nl = tt;      col0 = nl * 16; }
      else if (tt < 20) { pw = P.pWq; bias = P.m_bq; nl = tt - 16; col0 = 256 + nl * 16; }
      else if (tt < 24) { pw = P.pWk; bias = P.m_bk; nl = tt - 20; col0 = 320 + nl * 16; }
      else              { pw = P.pWv; bias = P.m_bv; nl = tt - 24; col0 = 384 + nl * 16; }
      f32x8 acc; float bv = bias[nl * 16 + ln];
#pragma unroll
      for (int v = 0; v < 8; ++v) acc[v] = bv;
      const __bf16* ax = x1b + ln * AST;
#pragma unroll
      for (int kt = 0; kt < 8; ++kt)
        acc = wmma_bf16(load_a_lds(ax, kt * 32, half8),
                        load_b_packed(pw, nl * 8 + kt, lane), acc);
#pragma unroll
      for (int v = 0; v < 8; ++v) gates[(v + half8) * GST + col0 + ln] = acc[v];
    }
    {  // i/f gate dot products: 16 rows x 16 segments of 16
      int m = tid >> 4, seg = tid & 15;
      float si = 0.f, sf = 0.f;
      for (int c = 0; c < 16; ++c) {
        float xv = bf2f(x1b[m * AST + seg * 16 + c]);
        si += xv * P.m_Wi[seg * 16 + c];
        sf += xv * P.m_Wf[seg * 16 + c];
      }
      atomicAdd(&ipre[m], si);
      atomicAdd(&fpre[m], sf);
    }
    __syncthreads();

    // ---- B2: i1/f1/n1 + o1 = sigmoid in place ----
    if (tid < 16) {
      float i1 = __expf(ipre[tid]), f1 = sgm(fpre[tid]);
      i1a[tid] = i1; f1a[tid] = f1;
      n1a[tid] = f1 * n1a[tid] + i1;
    }
    for (int r = tid; r < 4096; r += 256) {
      int m = r >> 8, h = r & 255;
      gates[m * GST + h] = sgm(gates[m * GST + h]);
    }
    __syncthreads();

    // ---- Part C: C = f*C + i*(v x k) ; htilde = (C@q)/(n1+eps) ----
    for (int p = tid; p < 1024; p += 256) {
      int m = p >> 6, i = p & 63;
      float f1 = f1a[m], vv = i1a[m] * gates[m * GST + 384 + i];
      float* Crow = P.C1 + (((size_t)(bg * 16 + m) * 64 + i) * 64);
      const float* qrow = gates + m * GST + 256;
      const float* krow = gates + m * GST + 320;
      float acc = 0.f;
#pragma unroll 4
      for (int j4 = 0; j4 < 16; ++j4) {
        float4 c = *(float4*)(Crow + j4 * 4);
        float4 k = *(const float4*)(krow + j4 * 4);
        float4 q = *(const float4*)(qrow + j4 * 4);
        c.x = f1 * c.x + vv * k.x; c.y = f1 * c.y + vv * k.y;
        c.z = f1 * c.z + vv * k.z; c.w = f1 * c.w + vv * k.w;
        acc += c.x * q.x + c.y * q.y + c.z * q.z + c.w * q.w;
        *(float4*)(Crow + j4 * 4) = c;
      }
      htb[m * 72 + i] = f2bf(acc / (n1a[m] + EPSF));
    }
    __syncthreads();

    // ---- Part D: h1 = o1*tanh(htilde@Wp + bp) ; x2 = h1 + x1 ----
    for (int tt = wave; tt < 16; tt += 8) {
      f32x8 acc; float bv = P.m_bp[tt * 16 + ln];
#pragma unroll
      for (int v = 0; v < 8; ++v) acc[v] = bv;
      const __bf16* ax = htb + ln * 72;
#pragma unroll
      for (int kt = 0; kt < 2; ++kt)
        acc = wmma_bf16(load_a_lds(ax, kt * 32, half8),
                        load_b_packed(P.pWp, tt * 2 + kt, lane), acc);
#pragma unroll
      for (int v = 0; v < 8; ++v) {
        int mm = v + half8, col = tt * 16 + ln;
        float h1 = gates[mm * GST + col] * tanhf(acc[v]);
        x2b[mm * AST + col] = f2bf(h1 + bf2f(x1b[mm * AST + col]));
      }
    }
    __syncthreads();

    // ---- Part E: subspace gates, 4 cells x 16 n-tiles, K=64 ----
    for (int tt = wave; tt < 64; tt += 8) {
      int s = tt >> 4, nt2 = tt & 15;
      f32x8 acc; float bv = P.sub_bih[s * 256 + nt2 * 16 + ln];
#pragma unroll
      for (int v = 0; v < 8; ++v) acc[v] = bv;
      const __bf16* ax = x2b + ln * AST + s * 64;
      const __bf16* ah = hsb + (s * 16 + ln) * 72;
      const __bf16* pwi = P.pSubWih + (size_t)s * 64 * 256;
      const __bf16* pwh = P.pSubWhh + (size_t)s * 64 * 256;
#pragma unroll
      for (int kt = 0; kt < 2; ++kt)
        acc = wmma_bf16(load_a_lds(ax, kt * 32, half8),
                        load_b_packed(pwi, nt2 * 2 + kt, lane), acc);
#pragma unroll
      for (int kt = 0; kt < 2; ++kt)
        acc = wmma_bf16(load_a_lds(ah, kt * 32, half8),
                        load_b_packed(pwh, nt2 * 2 + kt, lane), acc);
#pragma unroll
      for (int v = 0; v < 8; ++v)
        gates[(v + half8) * GST + s * 256 + nt2 * 16 + ln] = acc[v];
    }
    __syncthreads();

    // ---- E2: subspace sLSTM pointwise ----
    for (int r = tid; r < 4096; r += 256) {
      int s = r >> 10, m = (r >> 6) & 15, d = r & 63;
      float* g = gates + m * GST + s * 256;
      float i = __expf(g[d]);
      float f = sgm(g[64 + d] + P.sub_fbias[s * 64 + d]);
      float o = sgm(g[128 + d]);
      int idx = (s * 16 + m) * 64 + d;
      float cn = f * cs[idx] + i * tanhf(g[192 + d]);
      float nn = f * ns[idx] + i;
      cs[idx] = cn; ns[idx] = nn;
      hsb[(s * 16 + m) * 72 + d] = f2bf(o * tanhf(cn / (nn + EPSF)));
    }
    if (tid < 64) score[tid] = P.att_b2[0];
    __syncthreads();

    // ---- attention fusion + LN: only the last carry is observed ----
    if (t == Tsz - 1) {
      for (int tt = wave; tt < 16; tt += 8) {
        int s = tt >> 2, nt2 = tt & 3;
        f32x8 acc; float bv = P.att_b1[nt2 * 16 + ln];
#pragma unroll
        for (int v = 0; v < 8; ++v) acc[v] = bv;
        const __bf16* ax = hsb + (s * 16 + ln) * 72;
#pragma unroll
        for (int kt = 0; kt < 2; ++kt)
          acc = wmma_bf16(load_a_lds(ax, kt * 32, half8),
                          load_b_packed(P.pAttW1, nt2 * 2 + kt, lane), acc);
        float w2 = P.att_W2[nt2 * 16 + ln];
#pragma unroll
        for (int v = 0; v < 8; ++v)
          atomicAdd(&score[s * 16 + v + half8], tanhf(acc[v]) * w2);
      }
      __syncthreads();
      if (tid < 16) {  // softmax over S=4
        int m = tid;
        float mx = -1e30f;
        for (int s = 0; s < 4; ++s) mx = fmaxf(mx, score[s * 16 + m]);
        float e[4], sum = 0.f;
        for (int s = 0; s < 4; ++s) { e[s] = __expf(score[s * 16 + m] - mx); sum += e[s]; }
        for (int s = 0; s < 4; ++s) wsm[s * 16 + m] = e[s] / sum;
      }
      __syncthreads();
      for (int r = tid; r < 1024; r += 256) {
        int m = r >> 6, d = r & 63;
        float acc = 0.f;
        for (int s = 0; s < 4; ++s)
          acc += wsm[s * 16 + m] * bf2f(hsb[(s * 16 + m) * 72 + d]);
        hfu[m * 64 + d] = acc;
      }
      __syncthreads();
      if (tid < 16) {  // LN stats of tiled vector == stats over 64 values
        int m = tid;
        float mu = 0.f;
        for (int d = 0; d < 64; ++d) mu += hfu[m * 64 + d];
        mu *= (1.0f / 64.0f);
        float var = 0.f;
        for (int d = 0; d < 64; ++d) { float z = hfu[m * 64 + d] - mu; var += z * z; }
        var *= (1.0f / 64.0f);
        ipre[m] = mu; fpre[m] = rsqrtf(var + 1e-5f);
      }
      __syncthreads();
      for (int r = tid; r < 4096; r += 256) {
        int m = r >> 8, h = r & 255;
        float v = (hfu[m * 64 + (h & 63)] - ipre[m]) * fpre[m] * P.ln_g[h] + P.ln_b[h];
        P.out[(size_t)(bg * 16 + m) * Hsz + h] = v;
      }
    }
  }
}

// ---------------------------------------------------------------------------
extern "C" void kernel_launch(void* const* d_in, const int* in_sizes, int n_in,
                              void* d_out, int out_size, void* d_ws, size_t ws_size,
                              hipStream_t stream) {
  (void)in_sizes; (void)n_in; (void)out_size; (void)ws_size;
  const float* x       = (const float*)d_in[0];
  const float* W_in    = (const float*)d_in[1];
  const float* b_in    = (const float*)d_in[2];
  const float* s0_Wih  = (const float*)d_in[3];
  const float* s0_bih  = (const float*)d_in[4];
  const float* s0_Whh  = (const float*)d_in[5];
  const float* m_Wq    = (const float*)d_in[6];
  const float* m_bq    = (const float*)d_in[7];
  const float* m_Wk    = (const float*)d_in[8];
  const float* m_bk    = (const float*)d_in[9];
  const float* m_Wv    = (const float*)d_in[10];
  const float* m_bv    = (const float*)d_in[11];
  const float* m_Wi    = (const float*)d_in[12];
  const float* m_bi    = (const float*)d_in[13];
  const float* m_Wf    = (const float*)d_in[14];
  const float* m_bf    = (const float*)d_in[15];
  const float* m_Wo    = (const float*)d_in[16];
  const float* m_bo    = (const float*)d_in[17];
  const float* m_Wp    = (const float*)d_in[18];
  const float* m_bp    = (const float*)d_in[19];
  const float* sub_Wih = (const float*)d_in[20];
  const float* sub_bih = (const float*)d_in[21];
  const float* sub_Whh = (const float*)d_in[22];
  const float* sub_fb  = (const float*)d_in[23];
  const float* att_W1  = (const float*)d_in[24];
  const float* att_b1  = (const float*)d_in[25];
  const float* att_W2  = (const float*)d_in[26];
  const float* att_b2  = (const float*)d_in[27];
  const float* ln_g    = (const float*)d_in[28];
  const float* ln_b    = (const float*)d_in[29];

  // workspace carve (~70MB): xh bf16 + C f32 + packed bf16 weights
  char* ws = (char*)d_ws;
  size_t off = 0;
  __bf16* xh = (__bf16*)(ws + off); off += (size_t)Tsz * Bsz * Hsz * 2;
  float*  C1 = (float*)(ws + off);  off += (size_t)Bsz * HDsz * HDsz * 4;
  auto nb = [&](size_t bytes) { __bf16* p = (__bf16*)(ws + off); off += bytes; return p; };
  __bf16* pWin    = nb(128 * 256 * 2);
  __bf16* pWih0   = nb(256 * 1024 * 2);
  __bf16* pWhh0   = nb(256 * 1024 * 2);
  __bf16* pWq     = nb(256 * 64 * 2);
  __bf16* pWk     = nb(256 * 64 * 2);
  __bf16* pWv     = nb(256 * 64 * 2);
  __bf16* pWo     = nb(256 * 256 * 2);
  __bf16* pWp     = nb(64 * 256 * 2);
  __bf16* pSubWih = nb(4 * 64 * 256 * 2);
  __bf16* pSubWhh = nb(4 * 64 * 256 * 2);
  __bf16* pAttW1  = nb(64 * 64 * 2);

  auto pack = [&](const float* W, __bf16* dst, int K, int N) {
    int total = K * N;
    pack_b_kernel<<<(total + 255) / 256, 256, 0, stream>>>(W, dst, K, N);
  };
  pack(W_in, pWin, 128, 256);
  pack(s0_Wih, pWih0, 256, 1024);
  pack(s0_Whh, pWhh0, 256, 1024);
  pack(m_Wq, pWq, 256, 64);
  pack(m_Wk, pWk, 256, 64);
  pack(m_Wv, pWv, 256, 64);
  pack(m_Wo, pWo, 256, 256);
  pack(m_Wp, pWp, 64, 256);
  for (int s = 0; s < 4; ++s) {
    pack(sub_Wih + (size_t)s * 64 * 256, pSubWih + (size_t)s * 64 * 256, 64, 256);
    pack(sub_Whh + (size_t)s * 64 * 256, pSubWhh + (size_t)s * 64 * 256, 64, 256);
  }
  pack(att_W1, pAttW1, 64, 64);

  // phase 2: xh GEMM (131072 tiles, 8 waves/block)
  gemm_xh_kernel<<<(Tsz * 16 * 16) / 8, 256, 0, stream>>>(x, pWin, b_in, xh);

  // phase 3: persistent recurrent scan
  RecParams P;
  P.xh = xh; P.C1 = C1;
  P.pWih0 = pWih0; P.pWhh0 = pWhh0; P.pWq = pWq; P.pWk = pWk; P.pWv = pWv;
  P.pWo = pWo; P.pWp = pWp; P.pSubWih = pSubWih; P.pSubWhh = pSubWhh; P.pAttW1 = pAttW1;
  P.s0_bih = s0_bih; P.m_bq = m_bq; P.m_bk = m_bk; P.m_bv = m_bv;
  P.m_Wi = m_Wi; P.m_bi = m_bi; P.m_Wf = m_Wf; P.m_bf = m_bf;
  P.m_bo = m_bo; P.m_bp = m_bp;
  P.sub_bih = sub_bih; P.sub_fbias = sub_fb;
  P.att_b1 = att_b1; P.att_W2 = att_W2; P.att_b2 = att_b2;
  P.ln_g = ln_g; P.ln_b = ln_b;
  P.out = (float*)d_out;

  constexpr int SMEM =
      16 * GST * 4 + 4 * (16 * AST * 2) + 16 * 72 * 2 + 64 * 72 * 2 +
      2 * (16 * 256 * 4) + 2 * (64 * 64 * 4) + 16 * 64 * 4 +
      5 * 16 * 4 + 2 * 64 * 4;  // = 182,336 bytes (< 320KB/WGP on CDNA5)
  (void)hipFuncSetAttribute(reinterpret_cast<const void*>(recurrent_kernel),
                            hipFuncAttributeMaxDynamicSharedMemorySize, SMEM);
  recurrent_kernel<<<Bsz / 16, 256, SMEM, stream>>>(P);
}
